// GmxMACEModelNoPairs_25640954757414
// MI455X (gfx1250) — matrix operand order repaired and checked by reading, not compiled, not run on hardware
//
#include <hip/hip_runtime.h>
#include <math.h>

typedef __attribute__((ext_vector_type(16))) _Float16 v16h;
typedef __attribute__((ext_vector_type(4)))  _Float16 v4h;
typedef __attribute__((ext_vector_type(8)))  float    v8f;
typedef __attribute__((ext_vector_type(4)))  unsigned int u32x4;
typedef __attribute__((ext_vector_type(8)))  int      i32x8;
typedef __attribute__((ext_vector_type(4)))  int      i32x4;

#define N_NODES 50000
#define N_EDGES 1000000
#define C_CH    32
#define NBESS   8
#define NSPH    9
#define HID     64
#define RAD     (NSPH * C_CH)   /* 288 */
#define RMAXF   5.0f
#define LEN_CONV 10.0f
#define EN_CONV 96.4853f
#define PI_F    3.14159265358979f

#define AGG_BYTES ((size_t)N_NODES * RAD * sizeof(float)) /* 57.6 MB, 16B aligned */
#define TILE_H    512                                     /* f16 per 16x16-frag tile */

#if defined(__has_builtin)
#if __has_builtin(__builtin_amdgcn_tensor_load_to_lds) && \
    __has_builtin(__builtin_amdgcn_s_wait_tensorcnt)
#define HAVE_TDM 1
#endif
#endif

// ---------------------------------------------------------------------------
// WMMA helper (wave32, v_wmma_f32_16x16x32_f16)
// ---------------------------------------------------------------------------
__device__ __forceinline__ v8f wmma_f16(v16h a, v16h b, v8f c) {
  return __builtin_amdgcn_wmma_f32_16x16x32_f16(false, a, false, b, (short)0, c,
                                                false, false);
}

// Fragment-layout index helpers (inverse of ISA 7.12.2 wave32 layouts).
// A tile 16x32: element (m,k) -> lane m+((k>>3)&1)*16, pos ((k>>4)<<3)+(k&7)
__device__ __forceinline__ int a_idx(int m, int k) {
  return (m + (((k >> 3) & 1) << 4)) * 16 + (((k >> 4) << 3) + (k & 7));
}
// B tile 32x16: element (k,n) -> lane n+((k>>3)&1)*16, pos ((k>>4)<<3)+(k&7)
__device__ __forceinline__ int b_idx(int k, int n) {
  return (n + (((k >> 3) & 1) << 4)) * 16 + (((k >> 4) << 3) + (k & 7));
}

__device__ __forceinline__ float silu_f(float x) {
  return x * __builtin_amdgcn_rcpf(1.0f + __expf(-x));
}

#ifdef HAVE_TDM
// ---------------------------------------------------------------------------
// TDM: 1-D contiguous copy of n dwords global -> LDS via tensor_load_to_lds.
// D# per CDNA5 ISA: group0 {count=1, lds_addr, global_addr,
// type=2}; group1 {data_size=4B, tensor_dim0=tile_dim0=n, dim1=1, stride0=n}.
// Issue from one wave only (EXEC-independent DMA); tracked on TENSORcnt.
// ---------------------------------------------------------------------------
__device__ __forceinline__ void tdm_copy_to_lds(unsigned lds_off,
                                                const void* gsrc, unsigned n) {
  const unsigned long long ga = (unsigned long long)(uintptr_t)gsrc;
  u32x4 g0;
  g0[0] = 1u;                                   // count=1, user descriptor
  g0[1] = lds_off;                              // LDS byte address
  g0[2] = (unsigned)(ga & 0xFFFFFFFFu);         // global_addr[31:0]
  g0[3] = (unsigned)((ga >> 32) & 0x01FFFFFFu)  // global_addr[56:32]
          | (2u << 30);                         // type=2 ("image")
  i32x8 g1;
  g1[0] = (int)(2u << 16);                      // data_size=2 (4 bytes)
  g1[1] = (int)((n & 0xFFFFu) << 16);           // tensor_dim0[15:0]
  g1[2] = (int)((n >> 16) | (1u << 16));        // tensor_dim0[31:16], dim1=1
  g1[3] = (int)(n << 16);                       // tile_dim0 = n (n < 65536)
  g1[4] = 1;                                    // tile_dim1 = 1
  g1[5] = (int)n;                               // tensor_dim0_stride[31:0]
  g1[6] = 0;
  g1[7] = 0;
  const i32x4 gz = {0, 0, 0, 0};
#if __clang_major__ >= 23
  const i32x8 gz8 = {0, 0, 0, 0, 0, 0, 0, 0};
  __builtin_amdgcn_tensor_load_to_lds(g0, g1, gz, gz, gz8, 0);
#else
  __builtin_amdgcn_tensor_load_to_lds(g0, g1, gz, gz, 0);
#endif
}
#endif

// ---------------------------------------------------------------------------
// Kernel P: one-time conversion of weights to f16 in WMMA fragment layout.
// ---------------------------------------------------------------------------
__global__ void prep_weights(const float* __restrict__ W1,
                             const float* __restrict__ W2,
                             const float* __restrict__ W_mix,
                             const float* __restrict__ W_r1,
                             _Float16* __restrict__ W1f,
                             _Float16* __restrict__ W2f,
                             _Float16* __restrict__ Wmixf,
                             _Float16* __restrict__ Wr1f) {
  const int i0 = blockIdx.x * blockDim.x + threadIdx.x;
  const int stride = gridDim.x * blockDim.x;
  for (int idx = i0; idx < 32 * HID; idx += stride) {
    const int k = idx >> 6, col = idx & 63;
    const int nt = col >> 4, n = col & 15;
    const float v = (k < NBESS) ? W1[k * HID + col] : 0.0f;
    W1f[nt * TILE_H + b_idx(k, n)] = (_Float16)v;
  }
  for (int idx = i0; idx < HID * RAD; idx += stride) {
    const int k = idx / RAD, j = idx % RAD;
    const int kc = k >> 5, kk = k & 31;
    const int nt = j >> 4, n = j & 15;
    W2f[(nt * 2 + kc) * TILE_H + b_idx(kk, n)] = (_Float16)W2[idx];
  }
  for (int idx = i0; idx < RAD * C_CH; idx += stride) {
    const int k = idx >> 5, d = idx & 31;
    const int kc = k >> 5, kk = k & 31;
    const int nt = d >> 4, n = d & 15;
    Wmixf[(nt * 9 + kc) * TILE_H + b_idx(kk, n)] = (_Float16)W_mix[idx];
  }
  for (int idx = i0; idx < C_CH * HID; idx += stride) {
    const int k = idx >> 6, col = idx & 63;
    const int nt = col >> 4, n = col & 15;
    Wr1f[nt * TILE_H + b_idx(k, n)] = (_Float16)W_r1[idx];
  }
}

// ---------------------------------------------------------------------------
// Kernel 0: zero the agg workspace (L2-resident accumulator) and d_out
// ---------------------------------------------------------------------------
__global__ void zero_kernel(float* __restrict__ agg, float* __restrict__ out,
                            long long n) {
  long long i = (long long)blockIdx.x * blockDim.x + threadIdx.x;
  const long long stride = (long long)gridDim.x * blockDim.x;
  for (; i < n; i += stride) agg[i] = 0.0f;
  if (blockIdx.x == 0 && threadIdx.x == 0) out[0] = 0.0f;
}

// ---------------------------------------------------------------------------
// Kernel 1: per-edge radial MLP (WMMA) + spherical modulation + scatter-add
//   256 threads = 8 waves; each wave owns 16 edges.
//   Weights (40 KB, pre-swizzled) staged LDS-side by one TDM DMA, overlapped
//   with the geometry/VALU phase.
// ---------------------------------------------------------------------------
__global__ __launch_bounds__(256, 1) void edge_kernel(
    const float* __restrict__ positions, const int* __restrict__ species,
    const int* __restrict__ edge_index, const float* __restrict__ shifts,
    const float* __restrict__ W_embed, const _Float16* __restrict__ gWf,
    float* __restrict__ agg) {
  __shared__ __align__(16) _Float16 sWf[(4 + 36) * TILE_H];  // W1f | W2f
  __shared__ __align__(16) _Float16 sBessF[8][TILE_H];       // A tile
  __shared__ __align__(16) _Float16 sHidF[8][2 * TILE_H];    // 2 A tiles (K=64)
  __shared__ float sYE[8][16][NSPH];                         // env * Y
  __shared__ float sH0[8][16][C_CH];
  __shared__ int   sRcv[8][16];
  __shared__ int   sSpec[8][16];

  const _Float16* sW1f = sWf;
  const _Float16* sW2f = sWf + 4 * TILE_H;

  const int tid  = threadIdx.x;
  const int lane = tid & 31;
  const int wave = tid >> 5;
  const int hb   = (lane >> 4) << 3;  // D-fragment row base
  const int ncol = lane & 15;

#ifdef HAVE_TDM
  if (wave == 0) {
    tdm_copy_to_lds((unsigned)(uintptr_t)&sWf[0], gWf,
                    (40u * TILE_H * 2u) / 4u);  // 10240 dwords
  }
#else
  for (int i = tid; i < (40 * TILE_H) / 8; i += 256)
    ((uint4*)sWf)[i] = ((const uint4*)gWf)[i];
#endif

  // Zero bessel A-tile K-padding: each lane owns its fragment row exclusively.
  {
    const uint4 z = {0u, 0u, 0u, 0u};
    *(uint4*)&sBessF[wave][lane * 16 + 8] = z;                // K 16..31 band
    if (lane >= 16) *(uint4*)&sBessF[wave][lane * 16] = z;    // K 8..15 band
  }

  const int ebase = blockIdx.x * 128 + wave * 16;

  // Geometry: lanes 0..15, one edge each (overlaps with the weight DMA).
  if (lane < 16) {
    const int e = ebase + lane;
    float env = 0.0f;
    int rcv = 0, spec_snd = 0;
    float Y[NSPH];
    float bess[NBESS];
#pragma unroll
    for (int k = 0; k < NSPH; ++k) Y[k] = 0.0f;
#pragma unroll
    for (int k = 0; k < NBESS; ++k) bess[k] = 0.0f;
    if (e < N_EDGES) {
      const int snd = edge_index[e];
      rcv = edge_index[N_EDGES + e];
      const float vx = (positions[3 * rcv + 0] - positions[3 * snd + 0] + shifts[3 * e + 0]) * LEN_CONV;
      const float vy = (positions[3 * rcv + 1] - positions[3 * snd + 1] + shifts[3 * e + 1]) * LEN_CONV;
      const float vz = (positions[3 * rcv + 2] - positions[3 * snd + 2] + shifts[3 * e + 2]) * LEN_CONV;
      const float r   = sqrtf(vx * vx + vy * vy + vz * vz);
      const float rs  = fmaxf(r, 1e-6f);
      const float inv = __builtin_amdgcn_rcpf(rs);
      const float x = vx * inv, y = vy * inv, z = vz * inv;
      env = (r < RMAXF) ? 0.5f * (__cosf(PI_F * r / RMAXF) + 1.0f) : 0.0f;
      Y[0] = 1.0f; Y[1] = x; Y[2] = y; Y[3] = z;
      Y[4] = x * y; Y[5] = y * z; Y[6] = 3.0f * z * z - 1.0f;
      Y[7] = x * z; Y[8] = x * x - y * y;
      const float pref = sqrtf(2.0f / RMAXF) * inv;
#pragma unroll
      for (int n = 1; n <= NBESS; ++n)
        bess[n - 1] = pref * __sinf((float)n * PI_F * rs / RMAXF);
      spec_snd = species[snd];
    }
    sRcv[wave][lane]  = rcv;
    sSpec[wave][lane] = spec_snd;
#pragma unroll
    for (int k = 0; k < NSPH; ++k) sYE[wave][lane][k] = env * Y[k];
    union { _Float16 h[8]; uint4 u; } pk;
#pragma unroll
    for (int k = 0; k < NBESS; ++k) pk.h[k] = (_Float16)bess[k];
    *(uint4*)&sBessF[wave][lane * 16] = pk.u;
  }
  __syncthreads();

  // Gather sender embeddings (all 32 lanes).
  for (int i = lane; i < 16 * C_CH; i += 32) {
    const int m = i >> 5, c = i & 31;
    sH0[wave][m][c] = W_embed[sSpec[wave][m] * C_CH + c];
  }
#ifdef HAVE_TDM
  if (wave == 0) __builtin_amdgcn_s_wait_tensorcnt(0);  // weights landed
#endif
  __syncthreads();

  // GEMM1: hidden = silu(bess @ W1)  -> 4 WMMAs, scatter f16 into A-frag layout.
  {
    const v16h a = *(const v16h*)&sBessF[wave][lane * 16];
#pragma unroll
    for (int nt = 0; nt < 4; ++nt) {
      const v16h b = *(const v16h*)&sW1f[nt * TILE_H + lane * 16];
      v8f acc = {};
      acc = wmma_f16(a, b, acc);
      const int col = nt * 16 + ncol;
      const int kc = col >> 5, kk = col & 31;
#pragma unroll
      for (int i = 0; i < 8; ++i) {
        const int m = i + hb;
        sHidF[wave][kc * TILE_H + a_idx(m, kk)] = (_Float16)silu_f(acc[i]);
      }
    }
  }
  __syncthreads();

  // GEMM2: radial = hidden @ W2 (18 n-tiles x 2 K-chunks = 36 WMMAs)
  // epilogue: msg = radial * (env*Y[k]) * h0[snd,c] -> atomicAdd agg[rcv] (L2).
  {
    const v16h a0 = *(const v16h*)&sHidF[wave][0 * TILE_H + lane * 16];
    const v16h a1 = *(const v16h*)&sHidF[wave][1 * TILE_H + lane * 16];
    int rbase[8];
#pragma unroll
    for (int i = 0; i < 8; ++i) rbase[i] = sRcv[wave][i + hb] * RAD;
#pragma unroll
    for (int nt = 0; nt < 18; ++nt) {
      v8f acc = {};
      acc = wmma_f16(a0, *(const v16h*)&sW2f[(nt * 2 + 0) * TILE_H + lane * 16], acc);
      acc = wmma_f16(a1, *(const v16h*)&sW2f[(nt * 2 + 1) * TILE_H + lane * 16], acc);
      const int j  = nt * 16 + ncol;  // 0..287 = k*32 + c
      const int ks = j >> 5;
      const int c  = j & 31;
#pragma unroll
      for (int i = 0; i < 8; ++i) {
        const int m = i + hb;
        const float val = acc[i] * sYE[wave][m][ks] * sH0[wave][m][c];
        atomicAdd(&agg[rbase[i] + j], val);
      }
    }
  }
}

// ---------------------------------------------------------------------------
// Kernel 2: per-node mixing GEMM + readout MLP + energy reduction
//   256 threads = 8 waves; each wave owns 16 nodes. Weights via one TDM DMA.
// ---------------------------------------------------------------------------
__global__ __launch_bounds__(256, 1) void node_kernel(
    const float* __restrict__ agg, const int* __restrict__ species,
    const float* __restrict__ W_embed, const float* __restrict__ E0,
    const _Float16* __restrict__ gWn, const float* __restrict__ w_r2,
    float* __restrict__ out) {
  __shared__ __align__(16) _Float16 sWn[(18 + 4) * TILE_H];  // Wmixf | Wr1f
  __shared__ float sWr2[HID];
  __shared__ __align__(16) _Float16 sAggF[8][9 * TILE_H];    // 9 A-tiles / wave
  __shared__ __align__(16) _Float16 sFeatF[8][TILE_H];
  __shared__ float sEner[8][16];
  __shared__ int   sSpecN[8][16];

  const _Float16* sWmixF = sWn;
  const _Float16* sWr1F  = sWn + 18 * TILE_H;

  const int tid  = threadIdx.x;
  const int lane = tid & 31;
  const int wave = tid >> 5;
  const int hb   = (lane >> 4) << 3;
  const int ncol = lane & 15;

#ifdef HAVE_TDM
  if (wave == 0) {
    tdm_copy_to_lds((unsigned)(uintptr_t)&sWn[0], gWn,
                    (22u * TILE_H * 2u) / 4u);  // 5632 dwords
  }
#else
  for (int i = tid; i < (22 * TILE_H) / 8; i += 256)
    ((uint4*)sWn)[i] = ((const uint4*)gWn)[i];
#endif
  for (int i = tid; i < HID; i += 256) sWr2[i] = w_r2[i];

  const int nbase = blockIdx.x * 128 + wave * 16;

  // Stage agg tile: float4 coalesced reads -> f16 fragment scatter (b64 stores)
  for (int id = lane; id < 16 * (RAD / 4); id += 32) {
    const int m = id / (RAD / 4);
    const int k0 = (id % (RAD / 4)) * 4;
    const int n = nbase + m;
    float4 v = {0.f, 0.f, 0.f, 0.f};
    if (n < N_NODES) v = ((const float4*)agg)[(n * RAD + k0) >> 2];
    const int kc = k0 >> 5, kk = k0 & 31;
    v4h p;
    p[0] = (_Float16)v.x; p[1] = (_Float16)v.y;
    p[2] = (_Float16)v.z; p[3] = (_Float16)v.w;
    *(v4h*)&sAggF[wave][kc * TILE_H + a_idx(m, kk)] = p;  // pos multiple of 4
  }
  if (lane < 16) {
    sEner[wave][lane] = 0.0f;
    const int n = nbase + lane;
    sSpecN[wave][lane] = (n < N_NODES) ? species[n] : 0;
  }
#ifdef HAVE_TDM
  if (wave == 0) __builtin_amdgcn_s_wait_tensorcnt(0);
#endif
  __syncthreads();

  // feat = agg @ W_mix + h0 : kc-outer, 2 accumulators (A frag loaded once).
  {
    v8f acc0 = {}, acc1 = {};
#pragma unroll
    for (int kc = 0; kc < 9; ++kc) {
      const v16h a  = *(const v16h*)&sAggF[wave][kc * TILE_H + lane * 16];
      const v16h b0 = *(const v16h*)&sWmixF[(0 * 9 + kc) * TILE_H + lane * 16];
      const v16h b1 = *(const v16h*)&sWmixF[(1 * 9 + kc) * TILE_H + lane * 16];
      acc0 = wmma_f16(a, b0, acc0);
      acc1 = wmma_f16(a, b1, acc1);
    }
#pragma unroll
    for (int i = 0; i < 8; ++i) {
      const int m = i + hb;
      const int c0 = ncol, c1 = 16 + ncol;
      const float h0a = W_embed[sSpecN[wave][m] * C_CH + c0];
      const float h0b = W_embed[sSpecN[wave][m] * C_CH + c1];
      sFeatF[wave][a_idx(m, c0)] = (_Float16)(acc0[i] + h0a);
      sFeatF[wave][a_idx(m, c1)] = (_Float16)(acc1[i] + h0b);
    }
  }
  __syncthreads();

  // energy = silu(feat @ W_r1) . w_r2  (4 WMMAs) -> per-node scalar
  {
    const v16h a = *(const v16h*)&sFeatF[wave][lane * 16];
    float er[8];
#pragma unroll
    for (int i = 0; i < 8; ++i) er[i] = 0.0f;
#pragma unroll
    for (int nt = 0; nt < 4; ++nt) {
      const v16h b = *(const v16h*)&sWr1F[nt * TILE_H + lane * 16];
      v8f acc = {};
      acc = wmma_f16(a, b, acc);
      const float w = sWr2[nt * 16 + ncol];
#pragma unroll
      for (int i = 0; i < 8; ++i) er[i] += silu_f(acc[i]) * w;
    }
#pragma unroll
    for (int i = 0; i < 8; ++i) atomicAdd(&sEner[wave][i + hb], er[i]);
  }
  __syncthreads();

  if (lane < 16) {
    const int n = nbase + lane;
    if (n < N_NODES) {
      const float e = sEner[wave][lane] + E0[sSpecN[wave][lane]];
      atomicAdd(out, e * EN_CONV);
    }
  }
}

// ---------------------------------------------------------------------------
extern "C" void kernel_launch(void* const* d_in, const int* in_sizes, int n_in,
                              void* d_out, int out_size, void* d_ws,
                              size_t ws_size, hipStream_t stream) {
  const float* positions  = (const float*)d_in[0];
  const int*   species    = (const int*)d_in[1];
  const int*   edge_index = (const int*)d_in[2];
  const float* shifts     = (const float*)d_in[3];
  const float* W_embed    = (const float*)d_in[4];
  const float* E0         = (const float*)d_in[5];
  const float* W1         = (const float*)d_in[6];
  const float* W2         = (const float*)d_in[7];
  const float* W_mix      = (const float*)d_in[8];
  const float* W_r1       = (const float*)d_in[9];
  const float* w_r2       = (const float*)d_in[10];
  float* out = (float*)d_out;

  char* wsb = (char*)d_ws;
  float*    agg   = (float*)wsb;                 // 57.6 MB (L2-resident)
  _Float16* W1f   = (_Float16*)(wsb + AGG_BYTES);
  _Float16* W2f   = W1f + 4 * TILE_H;            // contiguous after W1f
  _Float16* Wmixf = W2f + 36 * TILE_H;
  _Float16* Wr1f  = Wmixf + 18 * TILE_H;         // contiguous after Wmixf

  prep_weights<<<64, 256, 0, stream>>>(W1, W2, W_mix, W_r1, W1f, W2f, Wmixf, Wr1f);

  const long long nz = (long long)N_NODES * RAD;
  zero_kernel<<<2048, 256, 0, stream>>>(agg, out, nz);

  const int eblocks = (N_EDGES + 127) / 128;
  edge_kernel<<<eblocks, 256, 0, stream>>>(positions, species, edge_index,
                                           shifts, W_embed, W1f, agg);

  const int nblocks = (N_NODES + 127) / 128;
  node_kernel<<<nblocks, 256, 0, stream>>>(agg, species, W_embed, E0, Wmixf,
                                           w_r2, out);
}
